// GRU_base2_60292750901498
// MI455X (gfx1250) — compile-verified, matrix operand order checked by reading, not run on hardware
//
#include <hip/hip_runtime.h>
#include <hip/hip_bf16.h>
#include <math.h>

// ---------------- problem constants (from reference) ----------------
constexpr int kB = 8, kS = 128, kN = 32, kE = 64;
constexpr int kD = 300, kHeads = 4, kF = 75, kHF = 300;
constexpr int kH = 512, kHS = 256;
constexpr int kG = kB * kS;          // 1024 graphs / sequence rows
constexpr int kVG = 50000, kVS = 25000;

// ---------------- WMMA types (gfx1250, wave32) ----------------
typedef __attribute__((ext_vector_type(16))) __bf16 v16bf;
typedef __attribute__((ext_vector_type(8)))  float  v8f;

union Frag {
  v16bf v;
  unsigned int u[8];
};

// round-to-nearest-even f32 -> bf16, packed pair into one dword
__device__ inline unsigned int pk_bf16(float a, float b) {
  unsigned int ua = __builtin_bit_cast(unsigned int, a);
  unsigned int ub = __builtin_bit_cast(unsigned int, b);
  ua += 0x7fffu + ((ua >> 16) & 1u);
  ub += 0x7fffu + ((ub >> 16) & 1u);
  return (ua >> 16) | (ub & 0xffff0000u);
}

// f32 source: pack 16 bf16 fragment elements (K = koff..+7, koff+16..+23)
__device__ inline void load_frag_f32(const float* __restrict__ base, int koff,
                                     unsigned int* u) {
  const float4* p0 = reinterpret_cast<const float4*>(base + koff);
  const float4* p1 = reinterpret_cast<const float4*>(base + koff + 16);
  float4 f0 = p0[0], f1 = p0[1];
  float4 f2 = p1[0], f3 = p1[1];
  u[0] = pk_bf16(f0.x, f0.y); u[1] = pk_bf16(f0.z, f0.w);
  u[2] = pk_bf16(f1.x, f1.y); u[3] = pk_bf16(f1.z, f1.w);
  u[4] = pk_bf16(f2.x, f2.y); u[5] = pk_bf16(f2.z, f2.w);
  u[6] = pk_bf16(f3.x, f3.y); u[7] = pk_bf16(f3.z, f3.w);
}

// bf16 source: two 16-byte vector loads, no packing VALU
__device__ inline void load_frag_bf16(const unsigned short* __restrict__ base,
                                      int koff, unsigned int* u) {
  const uint4* p0 = reinterpret_cast<const uint4*>(base + koff);
  const uint4* p1 = reinterpret_cast<const uint4*>(base + koff + 16);
  uint4 a = p0[0], b = p1[0];
  u[0] = a.x; u[1] = a.y; u[2] = a.z; u[3] = a.w;
  u[4] = b.x; u[5] = b.y; u[6] = b.z; u[7] = b.w;
}

// =====================================================================
// Kernel 1: fused gather + GAT (only node 0's aggregation is needed) +
// assembly of sig = [word_emb | global_state]  (one block per graph)
// =====================================================================
__global__ __launch_bounds__(256) void gat_kernel(
    const int* __restrict__ xi, const int* __restrict__ ei,
    const float* __restrict__ X, const float* __restrict__ gatW,
    const float* __restrict__ a_src, const float* __restrict__ a_dst,
    const float* __restrict__ gat_bias, float* __restrict__ sig) {
  __shared__ float hl[kN * kHF];            // 38.4 KB: h = x @ gatW
  __shared__ float s_asrc[kN * kHeads];
  __shared__ float s_adst[kN * kHeads];
  __shared__ float ev[(kE + 1) * kHeads];   // attention logits/weights to node 0
  __shared__ int   esrc[kE + 1];
  __shared__ float inv_den[kHeads];
  __shared__ int   nid[kN];

  const int g = blockIdx.x;
  const int tid = threadIdx.x;

  if (tid < kN) nid[tid] = xi[g * kN + tid];
  __syncthreads();

  // h[n, c] = sum_d X[nid[n], d] * gatW[d, c]   (X rows hit L0/L2)
  for (int o = tid; o < kN * kHF; o += 256) {
    int n = o / kHF, c = o - n * kHF;
    const float* xr = X + (size_t)nid[n] * kD;
    float acc = 0.f;
    for (int d = 0; d < kD; ++d) acc += xr[d] * gatW[d * kHF + c];
    hl[o] = acc;
  }
  __syncthreads();

  // asrc/adst: 32 nodes x 4 heads x {src,dst} = 256 tasks
  {
    int n = tid >> 3, hh = (tid >> 1) & 3, isdst = tid & 1;
    const float* av = (isdst ? a_dst : a_src) + hh * kF;
    const float* hr = hl + n * kHF + hh * kF;
    float acc = 0.f;
    for (int f = 0; f < kF; ++f) acc += hr[f] * av[f];
    if (isdst) s_adst[n * kHeads + hh] = acc;
    else       s_asrc[n * kHeads + hh] = acc;
  }
  __syncthreads();

  // attention logits for edges into node 0 (plus node-0 self loop at e == kE)
  for (int e = tid; e <= kE; e += 256) {
    int src = 0, dst = 0;
    if (e < kE) {
      src = ei[g * 2 * kE + e];
      dst = ei[g * 2 * kE + kE + e];
    }
    esrc[e] = src;
    for (int h = 0; h < kHeads; ++h) {
      float v = s_asrc[src * kHeads + h] + s_adst[h];   // adst of node 0
      v = (v > 0.f) ? v : 0.2f * v;                     // leaky_relu(0.2)
      ev[e * kHeads + h] = (dst == 0) ? v : -1e30f;
    }
  }
  __syncthreads();

  // per-head softmax over <=65 incoming edges (serial, tiny)
  if (tid < kHeads) {
    float m = -3.4e38f;
    for (int e = 0; e <= kE; ++e) m = fmaxf(m, ev[e * kHeads + tid]);
    float s = 0.f;
    for (int e = 0; e <= kE; ++e) {
      float w = expf(ev[e * kHeads + tid] - m);
      ev[e * kHeads + tid] = w;
      s += w;
    }
    inv_den[tid] = 1.0f / s;
  }
  __syncthreads();

  // global_state[c] = sum_e alpha_e * h[src_e, c]; sig = [word_emb | gs+bias]
  for (int c = tid; c < kHF; c += 256) {
    int h = c / kF;
    float acc = 0.f;
    for (int e = 0; e <= kE; ++e)
      acc += ev[e * kHeads + h] * hl[esrc[e] * kHF + c];
    sig[(size_t)g * 600 + 300 + c] = acc * inv_den[h] + gat_bias[c];
  }
  for (int d = tid; d < kD; d += 256)
    sig[(size_t)g * 600 + d] = X[(size_t)nid[0] * kD + d];
}

// =====================================================================
// Kernel 2: one GRU time step (thread = (batch b, unit j)); weights live
// in L2 (~7 MB), x_t and h_{t-1} staged in LDS.
// =====================================================================
__global__ __launch_bounds__(256) void gru_step(
    const float* __restrict__ xbase, int xstride,
    const float* __restrict__ hprev,
    const float* __restrict__ Wih, const float* __restrict__ Whh,
    const float* __restrict__ bih, const float* __restrict__ bhh,
    float* __restrict__ hout, int din, int hdim) {
  __shared__ float xs[kB * 600];
  __shared__ float hs[kB * kH];
  const int tid = threadIdx.x;

  for (int i = tid; i < kB * din; i += 256) {
    int b = i / din;
    xs[i] = xbase[(size_t)b * xstride + (i - b * din)];
  }
  for (int i = tid; i < kB * hdim; i += 256)
    hs[i] = hprev ? hprev[i] : 0.f;
  __syncthreads();

  const int gid = blockIdx.x * 256 + tid;
  const int b = gid & (kB - 1);
  const int j = gid >> 3;
  if (j >= hdim) return;

  const float* wr = Wih + (size_t)j * din;
  const float* wz = Wih + (size_t)(hdim + j) * din;
  const float* wn = Wih + (size_t)(2 * hdim + j) * din;
  const float* xb = xs + b * din;
  float ir = bih[j], iz = bih[hdim + j], in = bih[2 * hdim + j];
  for (int k = 0; k < din; ++k) {
    float x = xb[k];
    ir += x * wr[k]; iz += x * wz[k]; in += x * wn[k];
  }
  const float* ur = Whh + (size_t)j * hdim;
  const float* uz = Whh + (size_t)(hdim + j) * hdim;
  const float* un = Whh + (size_t)(2 * hdim + j) * hdim;
  const float* hb = hs + b * hdim;
  float hr = bhh[j], hz = bhh[hdim + j], hn = bhh[2 * hdim + j];
  for (int k = 0; k < hdim; ++k) {
    float h = hb[k];
    hr += h * ur[k]; hz += h * uz[k]; hn += h * un[k];
  }
  float r = 1.f / (1.f + expf(-(ir + hr)));
  float z = 1.f / (1.f + expf(-(iz + hz)));
  float n = tanhf(in + r * hn);
  float hp = hb[j];
  hout[b * hdim + j] = (1.f - z) * n + z * hp;
}

// =====================================================================
// Kernel 3a: elementwise f32 -> bf16 (RNE), 2 elements per thread-iter
// =====================================================================
__global__ __launch_bounds__(256) void cvt_bf16(const float* __restrict__ in,
                                                unsigned int* __restrict__ out,
                                                long npairs) {
  long stride = (long)gridDim.x * 256;
  for (long i = (long)blockIdx.x * 256 + threadIdx.x; i < npairs; i += stride) {
    const float2 f = reinterpret_cast<const float2*>(in)[i];
    out[i] = pk_bf16(f.x, f.y);
  }
}

// Kernel 3b: same, but remaps main-GRU rows (t,b,H) -> row m = b*S+t
__global__ __launch_bounds__(256) void cvt_bf16_remap(
    const float* __restrict__ in, unsigned int* __restrict__ out, long npairs) {
  constexpr int KP = kH / 2;  // pairs per row
  long stride = (long)gridDim.x * 256;
  for (long i = (long)blockIdx.x * 256 + threadIdx.x; i < npairs; i += stride) {
    int m = (int)(i / KP);
    int kp = (int)(i - (long)m * KP);
    size_t src = ((size_t)(((m & 127) << 3) + (m >> 7))) * kH + 2 * kp;
    out[i] = pk_bf16(in[src], in[src + 1]);
  }
}

// =====================================================================
// Kernel 4: C[M,V] = A[M,K] * W[V,K]^T + bias from pre-staged bf16.
// One wave -> 32x64 C strip: 2 A frags x 4 B frags -> 8 WMMAs per K-chunk.
// Out-of-range columns are CLAMPED (not predicated): loads stay legal and
// branch-free; garbage tiles are simply never stored. Inner loop is pure
// global_load_b128 + v_wmma.
// =====================================================================
__global__ __launch_bounds__(256) void gemm_wmma_pre(
    const unsigned short* __restrict__ A, const unsigned short* __restrict__ W,
    const float* __restrict__ bias, float* __restrict__ out,
    int M, int V, int K) {
  const int tid = threadIdx.x;
  const int lane = tid & 31;
  const int wave = blockIdx.x * 8 + (tid >> 5);
  const int mGroups = M >> 5;
  const int nGroups = (V + 63) >> 6;
  if (wave >= mGroups * nGroups) return;

  const int mi = wave % mGroups;
  const int gi = wave / mGroups;
  const int m0 = mi << 5;
  const int n0 = gi << 6;
  const int row = lane & 15;
  const int hi = lane >> 4;
  const int koff = hi << 3;

  const unsigned short* a0p = A + (size_t)(m0 + row) * K;
  const unsigned short* a1p = A + (size_t)(m0 + 16 + row) * K;

  const unsigned short* wp[4];
  for (int f = 0; f < 4; ++f) {
    int c = n0 + (f << 4) + row;
    int cc = (c < V) ? c : (V - 1);     // clamp: always a legal row
    wp[f] = W + (size_t)cc * K;
  }

  v8f acc0[4] = {v8f{}, v8f{}, v8f{}, v8f{}};
  v8f acc1[4] = {v8f{}, v8f{}, v8f{}, v8f{}};
  for (int kb = 0; kb < K; kb += 32) {
    Frag a0, a1;
    load_frag_bf16(a0p, kb + koff, a0.u);
    load_frag_bf16(a1p, kb + koff, a1.u);
    Frag bf[4];
    for (int f = 0; f < 4; ++f) load_frag_bf16(wp[f], kb + koff, bf[f].u);
    for (int f = 0; f < 4; ++f) {
      acc0[f] = __builtin_amdgcn_wmma_f32_16x16x32_bf16(
          false, a0.v, false, bf[f].v, (short)0, acc0[f], false, false);
      acc1[f] = __builtin_amdgcn_wmma_f32_16x16x32_bf16(
          false, a1.v, false, bf[f].v, (short)0, acc1[f], false, false);
    }
  }

  // D layout: VGPR v -> M = v + 8*hi, N = lane&15
  for (int f = 0; f < 4; ++f) {
    int c = n0 + (f << 4) + row;
    if (c < V) {
      float bv = bias[c];
      int mr0 = m0 + (hi << 3);
      for (int v = 0; v < 8; ++v)
        out[(size_t)(mr0 + v) * V + c] = acc0[f][v] + bv;
      int mr1 = m0 + 16 + (hi << 3);
      for (int v = 0; v < 8; ++v)
        out[(size_t)(mr1 + v) * V + c] = acc1[f][v] + bv;
    }
  }
}

// =====================================================================
// Kernel 4-fallback: on-the-fly f32->bf16 packing GEMM (used only if the
// workspace is too small for bf16 staging). amode==1 remaps b*S+t rows.
// Same column-clamp trick: branch-free inner loop.
// =====================================================================
__global__ __launch_bounds__(256) void gemm_wmma_fly(
    const float* __restrict__ A, int amode,
    const float* __restrict__ W, const float* __restrict__ bias,
    float* __restrict__ out, int M, int V, int K) {
  const int tid = threadIdx.x;
  const int lane = tid & 31;
  const int wave = blockIdx.x * 8 + (tid >> 5);
  const int mTiles = M >> 4;
  const int nGroups = (V + 63) >> 6;
  if (wave >= mTiles * nGroups) return;

  const int mi = wave % mTiles;
  const int gi = wave / mTiles;
  const int m0 = mi << 4;
  const int n0 = gi << 6;
  const int row = lane & 15;
  const int hi = lane >> 4;
  const int koff = hi << 3;

  const int m = m0 + row;
  const size_t abase = amode
      ? (size_t)(((m & 127) << 3) + (m >> 7)) * (size_t)K
      : (size_t)m * (size_t)K;

  size_t wbase[4];
  for (int f = 0; f < 4; ++f) {
    int c = n0 + (f << 4) + row;
    int cc = (c < V) ? c : (V - 1);
    wbase[f] = (size_t)cc * (size_t)K;
  }

  v8f acc[4] = {v8f{}, v8f{}, v8f{}, v8f{}};
  for (int kb = 0; kb < K; kb += 32) {
    Frag a;
    load_frag_f32(A + abase, kb + koff, a.u);
    Frag bf[4];
    for (int f = 0; f < 4; ++f)
      load_frag_f32(W + wbase[f], kb + koff, bf[f].u);
    for (int f = 0; f < 4; ++f) {
      acc[f] = __builtin_amdgcn_wmma_f32_16x16x32_bf16(
          false, a.v, false, bf[f].v, (short)0, acc[f], false, false);
    }
  }

  const int mrow = m0 + (hi << 3);
  for (int f = 0; f < 4; ++f) {
    int c = n0 + (f << 4) + row;
    if (c < V) {
      float bv = bias[c];
      for (int v = 0; v < 8; ++v)
        out[(size_t)(mrow + v) * V + c] = acc[f][v] + bv;
    }
  }
}

// =====================================================================
// Kernel 5: in-place row-wise log_softmax
// =====================================================================
__global__ __launch_bounds__(256) void log_softmax_kernel(float* __restrict__ x,
                                                          int V) {
  __shared__ float red[256];
  const int tid = threadIdx.x;
  float* p = x + (size_t)blockIdx.x * V;

  float m = -3.4e38f;
  for (int i = tid; i < V; i += 256) m = fmaxf(m, p[i]);
  red[tid] = m;
  __syncthreads();
  for (int s = 128; s; s >>= 1) {
    if (tid < s) red[tid] = fmaxf(red[tid], red[tid + s]);
    __syncthreads();
  }
  m = red[0];
  __syncthreads();

  float sum = 0.f;
  for (int i = tid; i < V; i += 256) sum += expf(p[i] - m);
  red[tid] = sum;
  __syncthreads();
  for (int s = 128; s; s >>= 1) {
    if (tid < s) red[tid] += red[tid + s];
    __syncthreads();
  }
  float lse = m + logf(red[0]);
  for (int i = tid; i < V; i += 256) p[i] -= lse;
}

// =====================================================================
extern "C" void kernel_launch(void* const* d_in, const int* in_sizes, int n_in,
                              void* d_out, int out_size, void* d_ws,
                              size_t ws_size, hipStream_t stream) {
  const int*   xi    = (const int*)  d_in[0];
  const int*   ei    = (const int*)  d_in[1];
  const float* X     = (const float*)d_in[2];
  const float* gatW  = (const float*)d_in[3];
  const float* asrc  = (const float*)d_in[4];
  const float* adst  = (const float*)d_in[5];
  const float* gbias = (const float*)d_in[6];
  const float* gWih0 = (const float*)d_in[7];
  const float* gWhh0 = (const float*)d_in[8];
  const float* gbih0 = (const float*)d_in[9];
  const float* gbhh0 = (const float*)d_in[10];
  const float* gWih1 = (const float*)d_in[11];
  const float* gWhh1 = (const float*)d_in[12];
  const float* gbih1 = (const float*)d_in[13];
  const float* gbhh1 = (const float*)d_in[14];
  const float* sWih0 = (const float*)d_in[15];
  const float* sWhh0 = (const float*)d_in[16];
  const float* sbih0 = (const float*)d_in[17];
  const float* sbhh0 = (const float*)d_in[18];
  const float* sWih1 = (const float*)d_in[19];
  const float* sWhh1 = (const float*)d_in[20];
  const float* sbih1 = (const float*)d_in[21];
  const float* sbhh1 = (const float*)d_in[22];
  const float* Wglob = (const float*)d_in[23];
  const float* bglob = (const float*)d_in[24];
  const float* Wsen  = (const float*)d_in[25];
  const float* bsen  = (const float*)d_in[26];

  // ---- workspace layout ----
  char* base = (char*)d_ws;
  float* sig   = (float*)base;                      // kG*600
  float* gout0 = sig   + (size_t)kG * 600;          // (t,b,H)
  float* gout1 = gout0 + (size_t)kS * kB * kH;
  float* sout0 = gout1 + (size_t)kS * kB * kH;      // (t,b,HS)
  float* sout1 = sout0 + (size_t)kS * kB * kHS;
  size_t floatBytes =
      ((size_t)kG * 600 + 2 * (size_t)kS * kB * kH + 2 * (size_t)kS * kB * kHS) *
      sizeof(float);

  unsigned short* Wg16 = (unsigned short*)(base + floatBytes);
  unsigned short* Ws16 = Wg16 + (size_t)kVG * kH;
  unsigned short* Ag16 = Ws16 + (size_t)kVS * kHS;
  unsigned short* As16 = Ag16 + (size_t)kG * kH;
  size_t needed = floatBytes +
      2 * ((size_t)kVG * kH + (size_t)kVS * kHS + (size_t)kG * kH +
           (size_t)kG * kHS);
  const bool staged = (ws_size >= needed);

  // stage weights in bf16 early (independent of GAT/GRU work)
  if (staged) {
    cvt_bf16<<<2048, 256, 0, stream>>>(Wglob, (unsigned int*)Wg16,
                                       (long)kVG * kH / 2);
    cvt_bf16<<<1024, 256, 0, stream>>>(Wsen, (unsigned int*)Ws16,
                                       (long)kVS * kHS / 2);
  }

  gat_kernel<<<kG, 256, 0, stream>>>(xi, ei, X, gatW, asrc, adst, gbias, sig);

  for (int t = 0; t < kS; ++t) {
    gru_step<<<(kB * kH) / 256, 256, 0, stream>>>(
        sig + (size_t)t * 600, kS * 600,
        t ? gout0 + (size_t)(t - 1) * kB * kH : (const float*)nullptr,
        gWih0, gWhh0, gbih0, gbhh0, gout0 + (size_t)t * kB * kH, 600, kH);
    gru_step<<<(kB * kH) / 256, 256, 0, stream>>>(
        gout0 + (size_t)t * kB * kH, kH,
        t ? gout1 + (size_t)(t - 1) * kB * kH : (const float*)nullptr,
        gWih1, gWhh1, gbih1, gbhh1, gout1 + (size_t)t * kB * kH, kH, kH);
    gru_step<<<(kB * kHS) / 256, 256, 0, stream>>>(
        sig + (size_t)t * 600, kS * 600,
        t ? sout0 + (size_t)(t - 1) * kB * kHS : (const float*)nullptr,
        sWih0, sWhh0, sbih0, sbhh0, sout0 + (size_t)t * kB * kHS, 600, kHS);
    gru_step<<<(kB * kHS) / 256, 256, 0, stream>>>(
        sout0 + (size_t)t * kB * kHS, kHS,
        t ? sout1 + (size_t)(t - 1) * kB * kHS : (const float*)nullptr,
        sWih1, sWhh1, sbih1, sbhh1, sout1 + (size_t)t * kB * kHS, kHS, kHS);
  }

  float* pg = (float*)d_out;
  float* ps = pg + (size_t)kG * kVG;

  if (staged) {
    cvt_bf16_remap<<<512, 256, 0, stream>>>(gout1, (unsigned int*)Ag16,
                                            (long)kG * kH / 2);
    cvt_bf16<<<256, 256, 0, stream>>>(sout1, (unsigned int*)As16,
                                      (long)kG * kHS / 2);
    {
      int nGroups = (kVG + 63) / 64;
      int waves = (kG / 32) * nGroups;
      gemm_wmma_pre<<<(waves + 7) / 8, 256, 0, stream>>>(Ag16, Wg16, bglob, pg,
                                                         kG, kVG, kH);
    }
    {
      int nGroups = (kVS + 63) / 64;
      int waves = (kG / 32) * nGroups;
      gemm_wmma_pre<<<(waves + 7) / 8, 256, 0, stream>>>(As16, Ws16, bsen, ps,
                                                         kG, kVS, kHS);
    }
  } else {
    {
      int nGroups = (kVG + 63) / 64;
      int waves = (kG / 16) * nGroups;
      gemm_wmma_fly<<<(waves + 7) / 8, 256, 0, stream>>>(
          gout1, /*amode=*/1, Wglob, bglob, pg, kG, kVG, kH);
    }
    {
      int nGroups = (kVS + 63) / 64;
      int waves = (kG / 16) * nGroups;
      gemm_wmma_fly<<<(waves + 7) / 8, 256, 0, stream>>>(
          sout1, /*amode=*/0, Wsen, bsen, ps, kG, kVS, kHS);
    }
  }

  log_softmax_kernel<<<kG, 256, 0, stream>>>(pg, kVG);
  log_softmax_kernel<<<kG, 256, 0, stream>>>(ps, kVS);
}
